// SlotAttention_4621384810957
// MI455X (gfx1250) — matrix operand order repaired and checked by reading, not compiled, not run on hardware
//
#include <hip/hip_runtime.h>

// ---------------------------------------------------------------------------
// Slot Attention for MI455X (gfx1250, wave32, WMMA).
// All large GEMMs run on v_wmma_f32_16x16x32_bf16; operands stored bf16 with
// fragment-friendly layouts read as k = 16h + s, not the ISA table order (A: row-major [M][K], B: row-major
// B^T [N][K], see CDNA5 ISA 7.12.2). Small ops (LN, softmax-over-8, GRU, MLP)
// are scalar wave32 kernels with LDS block reductions. Deterministic: split-K
// uses a partials buffer + reduce kernel, no float atomics.
// ---------------------------------------------------------------------------

#define B_ 64
#define N_ 4096
#define D_ 256
#define S_ 8
#define H_ 256
#define SCALE_ 0.0625f   // D^-0.5
#define EPS_ 1e-8f

typedef __attribute__((ext_vector_type(16))) __bf16 v16bf;
typedef __attribute__((ext_vector_type(8)))  __bf16 v8bf;
typedef __attribute__((ext_vector_type(8)))  float  v8f;

union AFrag { v16bf v; v8bf h[2]; };

static __device__ __forceinline__ v8f wmma_bf16(v16bf a, v16bf b, v8f c) {
  // (neg_a, A, neg_b, B, c_mod, C, reuse_a, reuse_b)
  return __builtin_amdgcn_wmma_f32_16x16x32_bf16(false, a, false, b, (short)0, c,
                                                 false, false);
}

static __device__ __forceinline__ float wave_sum(float v) {
#pragma unroll
  for (int m = 16; m >= 1; m >>= 1) v += __shfl_xor(v, m, 32);
  return v;
}

// 256-thread (8-wave) block sum; red must hold 8 floats of LDS.
static __device__ __forceinline__ float block_sum256(float v, float* red) {
  const int t = threadIdx.x;
  v = wave_sum(v);
  if ((t & 31) == 0) red[t >> 5] = v;
  __syncthreads();
  float tot = 0.f;
#pragma unroll
  for (int i = 0; i < 8; ++i) tot += red[i];
  __syncthreads();
  return tot;
}

// ---------------------------------------------------------------------------
// 0) convert Wk / Wv to bf16 once per launch
__global__ void wcvt_kernel(const float* __restrict__ wk, const float* __restrict__ wv,
                            __bf16* __restrict__ wkb, __bf16* __restrict__ wvb) {
  const int i = blockIdx.x * blockDim.x + threadIdx.x;
  if (i < D_ * D_) {
    wkb[i] = (__bf16)wk[i];
    wvb[i] = (__bf16)wv[i];
  }
}

// ---------------------------------------------------------------------------
// 1) LayerNorm(inputs) -> x_bf16 [B*N][D]; one wave per row (8 f32 per lane)
__global__ void ln_in_kernel(const float* __restrict__ in,
                             const float* __restrict__ g,
                             const float* __restrict__ b,
                             __bf16* __restrict__ xbf) {
  const size_t row  = (size_t)(blockIdx.x * blockDim.x + threadIdx.x) >> 5;
  const int    lane = threadIdx.x & 31;
  const float* rp   = in + row * D_;
  float vals[8];
  float s = 0.f;
#pragma unroll
  for (int i = 0; i < 8; ++i) { vals[i] = rp[lane * 8 + i]; s += vals[i]; }
  s = wave_sum(s);
  const float mean = s * (1.0f / D_);
  float s2 = 0.f;
#pragma unroll
  for (int i = 0; i < 8; ++i) { const float d = vals[i] - mean; s2 += d * d; }
  s2 = wave_sum(s2);
  const float inv = rsqrtf(s2 * (1.0f / D_) + 1e-5f);
  __bf16* op = xbf + row * D_;
#pragma unroll
  for (int i = 0; i < 8; ++i) {
    const int c = lane * 8 + i;
    op[c] = (__bf16)((vals[i] - mean) * inv * g[c] + b[c]);
  }
}

// ---------------------------------------------------------------------------
// 2) k = x@Wk^T + bk (bf16, [B*N][D]); v = x@Wv^T + bv stored transposed
//    vT [B][D][N] (bf16) so it serves as B^T for the updates GEMM.
//    Grid: (B*N)/16 blocks of 128 threads (4 waves); wave w does ntiles
//    {w, w+4, w+8, w+12}; A fragments (8 K-chunks) hoisted in registers.
__global__ void kv_gemm_kernel(const __bf16* __restrict__ xbf,
                               const __bf16* __restrict__ wk,
                               const __bf16* __restrict__ wv,
                               const float* __restrict__ bk,
                               const float* __restrict__ bv,
                               __bf16* __restrict__ kbf,
                               __bf16* __restrict__ vT) {
  const int mt   = blockIdx.x;
  const int wave = threadIdx.x >> 5;
  const int lane = threadIdx.x & 31;
  const int g    = lane >> 4;
  const int mr   = lane & 15;

  AFrag a[8];
  const __bf16* xrow = xbf + ((size_t)mt * 16 + mr) * D_;
#pragma unroll
  for (int kc = 0; kc < 8; ++kc) {
    a[kc].v = *(const v16bf*)(xrow + kc * 32 + g * 16);
  }

  for (int t = 0; t < 4; ++t) {
    const int nt = wave + t * 4;
    const __bf16* wkrow = wk + (size_t)(nt * 16 + mr) * D_;   // B^T row = Wk row
    const __bf16* wvrow = wv + (size_t)(nt * 16 + mr) * D_;
    v8f acck = {};
    v8f accv = {};
#pragma unroll
    for (int kc = 0; kc < 8; ++kc) {
      const v16bf bkf = *(const v16bf*)(wkrow + kc * 32 + g * 16);
      const v16bf bvf = *(const v16bf*)(wvrow + kc * 32 + g * 16);
      acck = wmma_bf16(a[kc].v, bkf, acck);
      accv = wmma_bf16(a[kc].v, bvf, accv);
    }
    // D fragment: element (m = vv + 8*g, n = mr)
#pragma unroll
    for (int vv = 0; vv < 8; ++vv) {
      const int    m    = vv + 8 * g;
      const int    dcol = nt * 16 + mr;
      const size_t R    = (size_t)mt * 16 + m;
      kbf[R * D_ + dcol] = (__bf16)(acck[vv] + bk[dcol]);
      const size_t bb = R >> 12;         // row / 4096
      const size_t j  = R & (N_ - 1);
      vT[(bb * D_ + dcol) * (size_t)N_ + j] = (__bf16)(accv[vv] + bv[dcol]);
    }
  }
}

// ---------------------------------------------------------------------------
// 3) slots = slots_mean + exp(slots_logvar) * slot_noise
__global__ void slots_init_kernel(const float* __restrict__ noise,
                                  const float* __restrict__ mean,
                                  const float* __restrict__ logvar,
                                  float* __restrict__ slots) {
  const int i = blockIdx.x * blockDim.x + threadIdx.x;
  const int d = i & (D_ - 1);
  slots[i] = mean[d] + __expf(logvar[d]) * noise[i];
}

// ---------------------------------------------------------------------------
// 4) sn = LN(slots); q = (sn@Wq^T + bq) * SCALE -> bf16 [B][16][D], rows 8..15
//    zero-padded (garbage-free A fragments for the dots GEMM).
__global__ void q_proj_kernel(const float* __restrict__ slots,
                              const float* __restrict__ gs,
                              const float* __restrict__ bs,
                              const float* __restrict__ wq,
                              const float* __restrict__ bq,
                              __bf16* __restrict__ qbf) {
  const int bsrow = blockIdx.x;              // b*8 + s
  const int t     = threadIdx.x;             // 0..255
  __shared__ float sn[D_];
  __shared__ float red[8];
  const float xv   = slots[(size_t)bsrow * D_ + t];
  const float mean = block_sum256(xv, red) * (1.0f / D_);
  const float diff = xv - mean;
  const float var  = block_sum256(diff * diff, red) * (1.0f / D_);
  const float inv  = rsqrtf(var + 1e-5f);
  sn[t] = diff * inv * gs[t] + bs[t];
  __syncthreads();
  float qv = bq[t];
  const float* wrow = wq + (size_t)t * D_;
  for (int c = 0; c < D_; ++c) qv += sn[c] * wrow[c];
  const int b = bsrow >> 3, s = bsrow & 7;
  qbf[((size_t)b * 16 + s) * D_ + t]     = (__bf16)(qv * SCALE_);
  qbf[((size_t)b * 16 + 8 + s) * D_ + t] = (__bf16)0.f;   // pad slot rows
}

// ---------------------------------------------------------------------------
// 5) dots[b][j][s] = SCALE * q[b,s,:]·k[b,j,:]  (SCALE folded into q).
//    Grid (64, 64) x 128 threads; wave w -> ntile = blockIdx.y*4+w.
//    Lanes 0..15 of the D fragment hold all 8 slots for column j -> v8f store.
__global__ void dots_gemm_kernel(const __bf16* __restrict__ qbf,
                                 const __bf16* __restrict__ kbf,
                                 float* __restrict__ dots) {
  const int b    = blockIdx.x;
  const int wave = threadIdx.x >> 5;
  const int lane = threadIdx.x & 31;
  const int g    = lane >> 4;
  const int mr   = lane & 15;
  const int nt   = blockIdx.y * 4 + wave;

  const __bf16* qrow = qbf + ((size_t)b * 16 + mr) * D_;
  const __bf16* krow = kbf + ((size_t)b * N_ + nt * 16 + mr) * D_;  // B^T rows
  v8f acc = {};
#pragma unroll
  for (int kc = 0; kc < 8; ++kc) {
    AFrag a;
    a.v = *(const v16bf*)(qrow + kc * 32 + g * 16);
    const v16bf bf = *(const v16bf*)(krow + kc * 32 + g * 16);
    acc = wmma_bf16(a.v, bf, acc);
  }
  if (g == 0) {  // m = vv = slot 0..7, n = mr -> j = nt*16+mr
    *(v8f*)(dots + ((size_t)b * N_ + nt * 16 + mr) * 8) = acc;
  }
}

// ---------------------------------------------------------------------------
// 6) softmax over the 8 slots per (b,j), +EPS; write bf16 [B][16][N] (padded)
//    for the updates GEMM and f32 [B][S][N] into d_out's attn region.
__global__ void softmax_kernel(const float* __restrict__ dots,
                               __bf16* __restrict__ attn_bf,
                               float* __restrict__ attn_out) {
  const size_t idx = (size_t)blockIdx.x * blockDim.x + threadIdx.x;  // b*N + j
  const float* p = dots + idx * 8;
  float v[8];
  float mx = p[0];
#pragma unroll
  for (int i = 0; i < 8; ++i) { v[i] = p[i]; mx = fmaxf(mx, v[i]); }
  float sum = 0.f;
#pragma unroll
  for (int i = 0; i < 8; ++i) { v[i] = __expf(v[i] - mx); sum += v[i]; }
  const float inv = 1.0f / sum;
  const size_t b = idx >> 12, j = idx & (N_ - 1);
#pragma unroll
  for (int s = 0; s < 8; ++s) {
    const float a = v[s] * inv + EPS_;
    attn_bf[((b * 16 + s) << 12) + j]  = (__bf16)a;
    attn_out[((b * 8 + s) << 12) + j]  = a;
  }
#pragma unroll
  for (int s = 8; s < 16; ++s) attn_bf[((b * 16 + s) << 12) + j] = (__bf16)0.f;
}

// ---------------------------------------------------------------------------
// 7) updates partials: split-K over j (8 segments of 512). A = attn_bf rows,
//    B^T = vT rows. Raw D fragments dumped per (seg,b,ntile) -> deterministic.
__global__ void upd_part_kernel(const __bf16* __restrict__ attn_bf,
                                const __bf16* __restrict__ vT,
                                float* __restrict__ part) {
  const int seg  = blockIdx.x;
  const int b    = blockIdx.y;
  const int wave = threadIdx.x >> 5;
  const int lane = threadIdx.x & 31;
  const int g    = lane >> 4;
  const int mr   = lane & 15;
  const int nt   = blockIdx.z * 4 + wave;

  const __bf16* arow = attn_bf + ((size_t)b * 16 + mr) * N_;          // slot mr
  const __bf16* brow = vT + ((size_t)b * D_ + nt * 16 + mr) * N_;     // d column
  v8f acc = {};
  for (int kc = seg * 16; kc < seg * 16 + 16; ++kc) {
    AFrag a;
    a.v = *(const v16bf*)(arow + kc * 32 + g * 16);
    const v16bf bf = *(const v16bf*)(brow + kc * 32 + g * 16);
    acc = wmma_bf16(a.v, bf, acc);
  }
  *(v8f*)(part + (((size_t)seg * B_ + b) * 16 + nt) * 256 + lane * 8) = acc;
}

// 7b) reduce 8 segment partials, unpack fragment layout -> updates [B][8][D]
__global__ void upd_reduce_kernel(const float* __restrict__ part,
                                  float* __restrict__ updates) {
  const int bt = blockIdx.x;           // b*16 + ntile
  const int b  = bt >> 4, nt = bt & 15;
  const int f  = threadIdx.x;          // 0..255 raw fragment slot
  float s = 0.f;
#pragma unroll
  for (int seg = 0; seg < 8; ++seg)
    s += part[(((size_t)seg * B_ + b) * 16 + nt) * 256 + f];
  const int L = f >> 3, vv = f & 7;
  const int m = vv + 8 * (L >> 4);     // slot
  const int n = L & 15;                // d within tile
  if (m < S_) updates[((size_t)b * S_ + m) * D_ + nt * 16 + n] = s;
}

// ---------------------------------------------------------------------------
// 8) GRU cell + LN + MLP residual, one 256-thread block per (b,s) row.
__global__ void gru_mlp_kernel(const float* __restrict__ updates,
                               float* __restrict__ slots,
                               const float* __restrict__ w_ih, const float* __restrict__ b_ih,
                               const float* __restrict__ w_hh, const float* __restrict__ b_hh,
                               const float* __restrict__ gff,  const float* __restrict__ bff,
                               const float* __restrict__ w1,   const float* __restrict__ b1,
                               const float* __restrict__ w2,   const float* __restrict__ b2,
                               float* __restrict__ out_slots) {
  const int rs = blockIdx.x;     // b*8 + s
  const int t  = threadIdx.x;    // 0..255
  __shared__ float su[D_], shp[D_], sfb[D_], hid[H_];
  __shared__ float red[8];
  su[t]  = updates[(size_t)rs * D_ + t];
  shp[t] = slots[(size_t)rs * D_ + t];
  __syncthreads();

  float gir = b_ih[t], giz = b_ih[D_ + t], gin = b_ih[2 * D_ + t];
  float ghr = b_hh[t], ghz = b_hh[D_ + t], ghn = b_hh[2 * D_ + t];
  const float* wi0 = w_ih + (size_t)t * D_;
  const float* wi1 = w_ih + (size_t)(D_ + t) * D_;
  const float* wi2 = w_ih + (size_t)(2 * D_ + t) * D_;
  const float* wh0 = w_hh + (size_t)t * D_;
  const float* wh1 = w_hh + (size_t)(D_ + t) * D_;
  const float* wh2 = w_hh + (size_t)(2 * D_ + t) * D_;
  for (int c = 0; c < D_; ++c) {
    const float u = su[c], h = shp[c];
    gir += u * wi0[c]; giz += u * wi1[c]; gin += u * wi2[c];
    ghr += h * wh0[c]; ghz += h * wh1[c]; ghn += h * wh2[c];
  }
  const float r = 1.0f / (1.0f + __expf(-(gir + ghr)));
  const float z = 1.0f / (1.0f + __expf(-(giz + ghz)));
  const float n = tanhf(gin + r * ghn);
  const float h = (1.0f - z) * n + z * shp[t];
  __syncthreads();

  // LN(h) with g_ff / bias_ff
  const float mean = block_sum256(h, red) * (1.0f / D_);
  const float diff = h - mean;
  const float var  = block_sum256(diff * diff, red) * (1.0f / D_);
  const float inv  = rsqrtf(var + 1e-5f);
  sfb[t] = diff * inv * gff[t] + bff[t];
  __syncthreads();

  float hv = b1[t];
  const float* w1r = w1 + (size_t)t * D_;
  for (int c = 0; c < D_; ++c) hv += sfb[c] * w1r[c];
  hid[t] = fmaxf(hv, 0.f);
  __syncthreads();

  float delta = b2[t];
  const float* w2r = w2 + (size_t)t * H_;
  for (int c = 0; c < H_; ++c) delta += hid[c] * w2r[c];
  const float nv = h + delta;
  slots[(size_t)rs * D_ + t]     = nv;
  out_slots[(size_t)rs * D_ + t] = nv;   // last iteration's value persists
}

// ---------------------------------------------------------------------------
// workspace layout (bytes); total ~410 MB
static constexpr size_t OFF_XBF   = 0;                                   // [B*N][D] bf16
static constexpr size_t OFF_KBF   = OFF_XBF   + (size_t)B_ * N_ * D_ * 2;
static constexpr size_t OFF_VT    = OFF_KBF   + (size_t)B_ * N_ * D_ * 2;
static constexpr size_t OFF_WKBF  = OFF_VT    + (size_t)B_ * N_ * D_ * 2;
static constexpr size_t OFF_WVBF  = OFF_WKBF  + (size_t)D_ * D_ * 2;
static constexpr size_t OFF_QBF   = OFF_WVBF  + (size_t)D_ * D_ * 2;     // [B][16][D] bf16
static constexpr size_t OFF_ATTNB = OFF_QBF   + (size_t)B_ * 16 * D_ * 2; // [B][16][N] bf16
static constexpr size_t OFF_DOTS  = OFF_ATTNB + (size_t)B_ * 16 * N_ * 2; // [B][N][8] f32
static constexpr size_t OFF_UPART = OFF_DOTS  + (size_t)B_ * N_ * 8 * 4;  // [8][B][16][256] f32
static constexpr size_t OFF_UPD   = OFF_UPART + (size_t)8 * B_ * 16 * 256 * 4; // [B][8][D] f32
static constexpr size_t OFF_SLOTS = OFF_UPD   + (size_t)B_ * S_ * D_ * 4; // [B][8][D] f32

extern "C" void kernel_launch(void* const* d_in, const int* in_sizes, int n_in,
                              void* d_out, int out_size, void* d_ws, size_t ws_size,
                              hipStream_t stream) {
  (void)in_sizes; (void)n_in; (void)out_size; (void)ws_size;
  const float* inputs      = (const float*)d_in[0];
  const float* slot_noise  = (const float*)d_in[1];
  const float* slots_mean  = (const float*)d_in[2];
  const float* slots_logvar= (const float*)d_in[3];
  const float* Wq  = (const float*)d_in[4];
  const float* bq  = (const float*)d_in[5];
  const float* Wk  = (const float*)d_in[6];
  const float* bk  = (const float*)d_in[7];
  const float* Wv  = (const float*)d_in[8];
  const float* bv  = (const float*)d_in[9];
  const float* W_ih = (const float*)d_in[10];
  const float* b_ih = (const float*)d_in[11];
  const float* W_hh = (const float*)d_in[12];
  const float* b_hh = (const float*)d_in[13];
  const float* W1  = (const float*)d_in[14];
  const float* b1  = (const float*)d_in[15];
  const float* W2  = (const float*)d_in[16];
  const float* b2  = (const float*)d_in[17];
  const float* g_in       = (const float*)d_in[18];
  const float* bias_in    = (const float*)d_in[19];
  const float* g_slots    = (const float*)d_in[20];
  const float* bias_slots = (const float*)d_in[21];
  const float* g_ff       = (const float*)d_in[22];
  const float* bias_ff    = (const float*)d_in[23];

  char* ws = (char*)d_ws;
  __bf16* xbf     = (__bf16*)(ws + OFF_XBF);
  __bf16* kbf     = (__bf16*)(ws + OFF_KBF);
  __bf16* vT      = (__bf16*)(ws + OFF_VT);
  __bf16* wkb     = (__bf16*)(ws + OFF_WKBF);
  __bf16* wvb     = (__bf16*)(ws + OFF_WVBF);
  __bf16* qbf     = (__bf16*)(ws + OFF_QBF);
  __bf16* attn_bf = (__bf16*)(ws + OFF_ATTNB);
  float*  dots    = (float*)(ws + OFF_DOTS);
  float*  upart   = (float*)(ws + OFF_UPART);
  float*  updates = (float*)(ws + OFF_UPD);
  float*  slots   = (float*)(ws + OFF_SLOTS);

  float* out_slots = (float*)d_out;                 // [B][S][D]
  float* out_attn  = out_slots + (size_t)B_ * S_ * D_; // [B][S][N]

  wcvt_kernel<<<(D_ * D_ + 255) / 256, 256, 0, stream>>>(Wk, Wv, wkb, wvb);
  ln_in_kernel<<<(B_ * N_) / 8, 256, 0, stream>>>(inputs, g_in, bias_in, xbf);
  kv_gemm_kernel<<<(B_ * N_) / 16, 128, 0, stream>>>(xbf, wkb, wvb, bk, bv, kbf, vT);
  slots_init_kernel<<<(B_ * S_ * D_) / 256, 256, 0, stream>>>(slot_noise, slots_mean,
                                                              slots_logvar, slots);

  for (int it = 0; it < 3; ++it) {
    q_proj_kernel<<<B_ * S_, 256, 0, stream>>>(slots, g_slots, bias_slots, Wq, bq, qbf);
    dots_gemm_kernel<<<dim3(B_, N_ / 64), 128, 0, stream>>>(qbf, kbf, dots);
    softmax_kernel<<<(B_ * N_) / 256, 256, 0, stream>>>(dots, attn_bf, out_attn);
    upd_part_kernel<<<dim3(8, B_, 4), 128, 0, stream>>>(attn_bf, vT, upart);
    upd_reduce_kernel<<<B_ * 16, 256, 0, stream>>>(upart, updates);
    gru_mlp_kernel<<<B_ * S_, 256, 0, stream>>>(updates, slots, W_ih, b_ih, W_hh, b_hh,
                                                g_ff, bias_ff, W1, b1, W2, b2, out_slots);
  }
}
